// MultiHeadAttention_34574486733535
// MI455X (gfx1250) — matrix operand order repaired and checked by reading, not compiled, not run on hardware
//
#include <hip/hip_runtime.h>
#include <hip/hip_bf16.h>

// ---------------------------------------------------------------------------
// Types for CDNA5 WMMA (wave32): 16x16x32 bf16 -> f32
// ---------------------------------------------------------------------------
typedef __attribute__((ext_vector_type(16))) __bf16 v16bf;
typedef __attribute__((ext_vector_type(8)))  __bf16 v8bf;
typedef __attribute__((ext_vector_type(4)))  __bf16 v4bf;
typedef __attribute__((ext_vector_type(8)))  float  v8f;
typedef int v4i __attribute__((vector_size(16)));   // matches builtin param type

#define MHA_B 4
#define MHA_T 2048
#define MHA_C 1024
#define MHA_H 16
#define MHA_D 64

#define GAS __attribute__((address_space(1)))
#define LAS __attribute__((address_space(3)))

#if __has_builtin(__builtin_amdgcn_global_load_async_to_lds_b128)
#define HAS_ASYNC_LDS 1
#endif

__device__ __forceinline__ v8bf ld8g(const __bf16* p) { return *(const v8bf*)p; }

__device__ __forceinline__ v16bf combine16(v8bf lo, v8bf hi) {
  return __builtin_shufflevector(lo, hi, 0,1,2,3,4,5,6,7,8,9,10,11,12,13,14,15);
}

// A fragment (16x32 bf16, M x K). lane: m = lane&15, hi = lane>>4.
// element i holds K = k0 + i + (i&8) + hi*8  -> two contiguous b128 loads.
__device__ __forceinline__ v16bf load_afrag(const __bf16* rowptr, int k0, int hi) {
  return combine16(ld8g(rowptr + k0 + hi * 8),
                   ld8g(rowptr + k0 + 16 + hi * 8));
}

// B fragment (32x16 bf16, K x N) sourced from Bt[n][k] (row-major, K contig).
// lane: n = lane&15, hi = lane>>4. element i holds K = k0 + i + (i&8) + hi*8.
__device__ __forceinline__ v16bf load_bfrag(const __bf16* colptr, int k0, int hi) {
  const __bf16* p = colptr + k0 + hi * 8;
  return combine16(ld8g(p), ld8g(p + 16));
}

__device__ __forceinline__ v8f wmma_bf16(v16bf a, v16bf b, v8f c) {
  return __builtin_amdgcn_wmma_f32_16x16x32_bf16(
      /*neg_a=*/false, a, /*neg_b=*/false, b,
      /*c_mod=*/(short)0, c, /*reuse_a=*/false, /*reuse_b=*/false);
}

// 16-byte global -> LDS copy: async path on gfx1250 (ASYNCcnt), ds fallback.
__device__ __forceinline__ void copy_b128_to_lds(const __bf16* g, __bf16* l) {
#ifdef HAS_ASYNC_LDS
  __builtin_amdgcn_global_load_async_to_lds_b128(
      (GAS v4i*)g, (LAS v4i*)l, /*imm_offset=*/0, /*cpol=*/0);
#else
  *(v8bf*)l = *(const v8bf*)g;
#endif
}

__device__ __forceinline__ void wait_async_lds() {
#if __has_builtin(__builtin_amdgcn_s_wait_asynccnt)
  __builtin_amdgcn_s_wait_asynccnt(0);
#elif defined(HAS_ASYNC_LDS)
  asm volatile("s_wait_asynccnt 0x0" ::: "memory");
#endif
}

// ---------------------------------------------------------------------------
// f32 -> bf16 convert, 4 elements / thread
// ---------------------------------------------------------------------------
__global__ __launch_bounds__(256) void cvt_f32_bf16x4(const float4* __restrict__ in,
                                                      v4bf* __restrict__ out, int n4) {
  int i = blockIdx.x * 256 + threadIdx.x;
  if (i < n4) {
    float4 v = in[i];
    v4bf o = {(__bf16)v.x, (__bf16)v.y, (__bf16)v.z, (__bf16)v.w};
    out[i] = o;
  }
}

// ---------------------------------------------------------------------------
// Transpose + convert: out[n][k] = (bf16) in[k][n], square side = side
// ---------------------------------------------------------------------------
__global__ __launch_bounds__(256) void transpose_cvt(const float* __restrict__ in,
                                                     __bf16* __restrict__ out, int side) {
  __shared__ __bf16 tile[32][33];
  int x = blockIdx.x * 32 + threadIdx.x;
#pragma unroll
  for (int j = threadIdx.y; j < 32; j += 8) {
    int y = blockIdx.y * 32 + j;
    tile[j][threadIdx.x] = (__bf16)in[(size_t)y * side + x];
  }
  __syncthreads();
  int xo = blockIdx.y * 32 + threadIdx.x;
#pragma unroll
  for (int j = threadIdx.y; j < 32; j += 8) {
    int yo = blockIdx.x * 32 + j;
    out[(size_t)yo * side + xo] = tile[threadIdx.x][j];
  }
}

// ---------------------------------------------------------------------------
// Fused GEMM: acc = A[M,K] @ Wt[N,K]^T + bias.
// Block = 256 threads = 8 waves; block tile 128(M) x 64(N); wave tile 16x64.
// B tile (64x32, shared by all 8 waves) is staged through double-buffered LDS
// via async load-to-LDS; A fragments are register-prefetched one k-step ahead.
// Epilogue modes: 0 -> bf16 [B,H,T,D]   (Q, K)
//                 1 -> bf16 [B,H,D,T]   (V transposed)
//                 2 -> f32  [M,N]       (final projection)
// ---------------------------------------------------------------------------
__device__ __forceinline__ void fill_btile(__bf16 (*buf)[32], const __bf16* Wt,
                                           int n0, int k0, int tid) {
  int n_local = tid >> 2;          // 0..63
  int c = (tid & 3) * 8;           // 0,8,16,24
  const __bf16* src = Wt + (size_t)(n0 + n_local) * MHA_C + k0 + c;
  copy_b128_to_lds(src, &buf[n_local][c]);
}

__global__ __launch_bounds__(256) void gemm_fused(const __bf16* __restrict__ Abf,
                                                  const __bf16* __restrict__ Wt,
                                                  const float* __restrict__ bias,
                                                  __bf16* __restrict__ outb,
                                                  float* __restrict__ outf,
                                                  int mode) {
  const int K = MHA_C, T = MHA_T, H = MHA_H, D = MHA_D;
  __shared__ __attribute__((aligned(16))) __bf16 btile[2][64][32];  // 8 KB

  int tid = threadIdx.x;
  int wave = tid >> 5, lane = tid & 31;
  int hi = lane >> 4, nl = lane & 15;
  int m0 = blockIdx.x * 128 + wave * 16;
  int n0 = blockIdx.y * 64;
  const __bf16* arow = Abf + (size_t)(m0 + nl) * K;

  // Prologue: stage B k-tile 0, prefetch A fragment 0.
  fill_btile(btile[0], Wt, n0, 0, tid);
  v16bf a = load_afrag(arow, 0, hi);
  wait_async_lds();
  __syncthreads();

  v8f acc[4] = {};
  for (int k0 = 0; k0 < K; k0 += 32) {
    int cur = (k0 >> 5) & 1;
    bool more = (k0 + 32 < K);
    if (more) fill_btile(btile[cur ^ 1], Wt, n0, k0 + 32, tid);  // async, in flight
    v16bf an = a;
    if (more) an = load_afrag(arow, k0 + 32, hi);                // reg prefetch

    v16bf b0 = load_bfrag(&btile[cur][0 + nl][0], 0, hi);
    v16bf b1 = load_bfrag(&btile[cur][16 + nl][0], 0, hi);
    v16bf b2 = load_bfrag(&btile[cur][32 + nl][0], 0, hi);
    v16bf b3 = load_bfrag(&btile[cur][48 + nl][0], 0, hi);
    acc[0] = wmma_bf16(a, b0, acc[0]);
    acc[1] = wmma_bf16(a, b1, acc[1]);
    acc[2] = wmma_bf16(a, b2, acc[2]);
    acc[3] = wmma_bf16(a, b3, acc[3]);

    a = an;
    wait_async_lds();   // next tile staged (had a full compute phase to land)
    __syncthreads();    // and everyone is done reading `cur`
  }

#pragma unroll
  for (int t = 0; t < 4; ++t) {
    int n = n0 + t * 16 + nl;
    float bv = bias[n];
    int h = n >> 6, d = n & (D - 1);
#pragma unroll
    for (int r = 0; r < 8; ++r) {
      int m = m0 + r + hi * 8;
      float val = acc[t][r] + bv;
      if (mode == 2) {
        outf[(size_t)m * MHA_C + n] = val;
      } else {
        int b = m >> 11, tok = m & (T - 1);
        size_t idx = (mode == 1) ? ((((size_t)(b * H + h)) * D + d) * T + tok)
                                 : ((((size_t)(b * H + h)) * T + tok) * D + d);
        outb[idx] = (__bf16)val;
      }
    }
  }
}

// ---------------------------------------------------------------------------
// Flash attention (causal). One wave owns a 16-row query tile with a 16x64
// f32 accumulator; streams 32 keys/step. Software pipelined: V fragments are
// loaded at iteration top (consumed only after softmax), next-step K
// fragments are prefetched during the softmax VALU phase.
// ---------------------------------------------------------------------------
__global__ __launch_bounds__(256) void flash_attn(const __bf16* __restrict__ Q,
                                                  const __bf16* __restrict__ Kc,
                                                  const __bf16* __restrict__ Vt,
                                                  __bf16* __restrict__ Y) {
  const int T = MHA_T, D = MHA_D, H = MHA_H, C = MHA_C;
  __shared__ __attribute__((aligned(16))) __bf16 plds[8][16][32];  // P staging

  int wave = threadIdx.x >> 5, lane = threadIdx.x & 31;
  int hi = lane >> 4, nl = lane & 15;
  int w = blockIdx.x * 8 + wave;
  int bh = w >> 7;                 // T/16 = 128 query tiles per (b,h)
  int q0 = (w & 127) << 4;

  const __bf16* Qp = Q + (size_t)bh * T * D;
  const __bf16* Kp = Kc + (size_t)bh * T * D;
  const __bf16* Vp = Vt + (size_t)bh * D * T;

  const __bf16* qrow = Qp + (size_t)(q0 + nl) * D;
  v16bf qa0 = load_afrag(qrow, 0, hi);
  v16bf qa1 = load_afrag(qrow, 32, hi);

  v8f acc[4] = {};
  float rmax[8], rsum[8];
#pragma unroll
  for (int r = 0; r < 8; ++r) { rmax[r] = -1e30f; rsum[r] = 0.0f; }
  const float scale = 0.125f;  // 1/sqrt(64)

  auto load_kfrags = [&](int s0, v16bf* kb) {
    const __bf16* r0 = Kp + (size_t)(s0 + nl) * D;
    const __bf16* r1 = Kp + (size_t)(s0 + 16 + nl) * D;
    kb[0] = load_bfrag(r0, 0, hi);
    kb[1] = load_bfrag(r0, 32, hi);
    kb[2] = load_bfrag(r1, 0, hi);
    kb[3] = load_bfrag(r1, 32, hi);
  };

  v16bf kb[4];
  load_kfrags(0, kb);

  for (int s0 = 0; s0 <= q0 + 15; s0 += 32) {
    // V fragments: issued now, consumed after softmax (latency hidden).
    v16bf vb[4];
#pragma unroll
    for (int t = 0; t < 4; ++t)
      vb[t] = load_bfrag(Vp + (size_t)(t * 16 + nl) * T + s0, 0, hi);

    // S = Q K^T for key groups [s0, s0+15] and [s0+16, s0+31]
    v8f sf0 = {}, sf1 = {};
    sf0 = wmma_bf16(qa0, kb[0], sf0);
    sf0 = wmma_bf16(qa1, kb[1], sf0);
    sf1 = wmma_bf16(qa0, kb[2], sf1);
    sf1 = wmma_bf16(qa1, kb[3], sf1);

    // Prefetch next-step K fragments during the softmax VALU phase.
    v16bf kbn[4] = {kb[0], kb[1], kb[2], kb[3]};
    if (s0 + 32 <= q0 + 15) load_kfrags(s0 + 32, kbn);

    // scale + causal mask
    bool need_mask = (s0 + 31 > q0);
    float p0[8], p1[8];
#pragma unroll
    for (int r = 0; r < 8; ++r) {
      float a0 = sf0[r] * scale, a1 = sf1[r] * scale;
      if (need_mask) {
        int tq = q0 + r + hi * 8;
        if (s0 + nl > tq)      a0 = -1e30f;
        if (s0 + 16 + nl > tq) a1 = -1e30f;
      }
      p0[r] = a0; p1[r] = a1;
    }

    // online softmax: reductions across the 16-lane halves
#pragma unroll
    for (int r = 0; r < 8; ++r) {
      float mx = fmaxf(p0[r], p1[r]);
#pragma unroll
      for (int o = 1; o < 16; o <<= 1) mx = fmaxf(mx, __shfl_xor(mx, o, 32));
      float mnew = fmaxf(rmax[r], mx);
      float corr = __expf(rmax[r] - mnew);
      float e0 = __expf(p0[r] - mnew);
      float e1 = __expf(p1[r] - mnew);
      float rs = e0 + e1;
#pragma unroll
      for (int o = 1; o < 16; o <<= 1) rs += __shfl_xor(rs, o, 32);
      rsum[r] = rsum[r] * corr + rs;
      rmax[r] = mnew;
      acc[0][r] *= corr; acc[1][r] *= corr;
      acc[2][r] *= corr; acc[3][r] *= corr;
      p0[r] = e0; p1[r] = e1;
    }

    // P: C-layout -> A-layout via per-wave LDS tile (in-order DS within wave)
#pragma unroll
    for (int r = 0; r < 8; ++r) {
      plds[wave][r + hi * 8][nl]      = (__bf16)p0[r];
      plds[wave][r + hi * 8][nl + 16] = (__bf16)p1[r];
    }
    __builtin_amdgcn_wave_barrier();
    v16bf pf = combine16(*(const v8bf*)&plds[wave][nl][hi * 8],
                         *(const v8bf*)&plds[wave][nl][16 + hi * 8]);
    __builtin_amdgcn_wave_barrier();

    // O += P @ V
#pragma unroll
    for (int t = 0; t < 4; ++t) acc[t] = wmma_bf16(pf, vb[t], acc[t]);

#pragma unroll
    for (int t = 0; t < 4; ++t) kb[t] = kbn[t];
  }

  // normalize and scatter into [B,T,C] bf16 for the final projection
  int b = bh >> 4, h = bh & (H - 1);
#pragma unroll
  for (int r = 0; r < 8; ++r) {
    float inv = 1.0f / rsum[r];
    int tq = q0 + r + hi * 8;
    size_t rowbase = ((size_t)b * T + tq) * C + h * D;
#pragma unroll
    for (int t = 0; t < 4; ++t)
      Y[rowbase + t * 16 + nl] = (__bf16)(acc[t][r] * inv);
  }
}

// ---------------------------------------------------------------------------
// Host launch
// ---------------------------------------------------------------------------
extern "C" void kernel_launch(void* const* d_in, const int* in_sizes, int n_in,
                              void* d_out, int out_size, void* d_ws, size_t ws_size,
                              hipStream_t stream) {
  const float* x  = (const float*)d_in[0];
  // d_in[1] = mask (causality handled analytically)
  const float* Wq = (const float*)d_in[2];
  const float* bq = (const float*)d_in[3];
  const float* Wk = (const float*)d_in[4];
  const float* bk = (const float*)d_in[5];
  const float* Wv = (const float*)d_in[6];
  const float* bv = (const float*)d_in[7];
  const float* Wp = (const float*)d_in[8];
  const float* bp = (const float*)d_in[9];
  float* out = (float*)d_out;

  const size_t M   = (size_t)MHA_B * MHA_T;   // 8192
  const size_t CC  = (size_t)MHA_C * MHA_C;   // 1M
  const size_t BTC = M * MHA_C;               // 8M elements

  char* ws = (char*)d_ws;
  __bf16* xbf = (__bf16*)ws; ws += BTC * 2;
  __bf16* wqt = (__bf16*)ws; ws += CC * 2;
  __bf16* wkt = (__bf16*)ws; ws += CC * 2;
  __bf16* wvt = (__bf16*)ws; ws += CC * 2;
  __bf16* wpt = (__bf16*)ws; ws += CC * 2;
  __bf16* qbf = (__bf16*)ws; ws += BTC * 2;
  __bf16* kbf = (__bf16*)ws; ws += BTC * 2;
  __bf16* vtb = (__bf16*)ws; ws += BTC * 2;
  __bf16* ybf = (__bf16*)ws; ws += BTC * 2;

  // 1) convert x, transpose+convert weights
  cvt_f32_bf16x4<<<(int)(BTC / 4 / 256), 256, 0, stream>>>(
      (const float4*)x, (v4bf*)xbf, (int)(BTC / 4));
  dim3 tb(32, 8), tg(MHA_C / 32, MHA_C / 32);
  transpose_cvt<<<tg, tb, 0, stream>>>(Wq, wqt, MHA_C);
  transpose_cvt<<<tg, tb, 0, stream>>>(Wk, wkt, MHA_C);
  transpose_cvt<<<tg, tb, 0, stream>>>(Wv, wvt, MHA_C);
  transpose_cvt<<<tg, tb, 0, stream>>>(Wp, wpt, MHA_C);

  // 2) Q/K/V projections (V written transposed [B,H,D,T])
  dim3 gg((unsigned)(M / 128), MHA_C / 64);
  gemm_fused<<<gg, 256, 0, stream>>>(xbf, wqt, bq, qbf, nullptr, 0);
  gemm_fused<<<gg, 256, 0, stream>>>(xbf, wkt, bk, kbf, nullptr, 0);
  gemm_fused<<<gg, 256, 0, stream>>>(xbf, wvt, bv, vtb, nullptr, 1);

  // 3) causal flash attention: B*H*(T/16) = 8192 wave-tiles, 8 waves/block
  flash_attn<<<1024, 256, 0, stream>>>(qbf, kbf, vtb, ybf);

  // 4) output projection (f32 out)
  gemm_fused<<<gg, 256, 0, stream>>>(ybf, wpt, bp, nullptr, out, 2);
}
